// GBottleneck_4466765988231
// MI455X (gfx1250) — matrix operand order repaired and checked by reading, not compiled, not run on hardware
//
#include <hip/hip_runtime.h>
#include <hip/hip_bf16.h>

#define NODES   50000
#define EDGES   300000
#define INDIM   963
#define HIDD    256
#define OUTD    3
#define NBLK    6
#define MPAD    50048          // 391 * 128  (rows padded so GEMM staging is guard-free)
#define PADK1   992            // 963 rounded up to a multiple of 32
#define TK      32
#define LDS_STRIDE 40          // ushorts per LDS row (32 data + 8 pad), 80 B = 16B-aligned rows

typedef __bf16 v16bf __attribute__((ext_vector_type(16)));
typedef float  v8f   __attribute__((ext_vector_type(8)));
typedef int    v4i   __attribute__((ext_vector_type(4)));

#if defined(__HIP_DEVICE_COMPILE__) && defined(__gfx1250__) && \
    __has_builtin(__builtin_amdgcn_global_load_async_to_lds_b128) && \
    __has_builtin(__builtin_amdgcn_s_wait_asynccnt)
#define USE_ASYNC 1
#else
#define USE_ASYNC 0
#endif

// ---------------------------------------------------------------------------
// float -> bf16 (round to nearest even) as raw ushort
// ---------------------------------------------------------------------------
__device__ __forceinline__ unsigned short f2bf(float f) {
    union { float f; unsigned u; } v; v.f = f;
    unsigned u = v.u;
    unsigned r = u + 0x7FFFu + ((u >> 16) & 1u);
    return (unsigned short)(r >> 16);
}

// 16-byte global -> LDS copy: async (ASYNCcnt) on gfx1250, plain b128 otherwise
__device__ __forceinline__ void copy16(const unsigned short* g, unsigned short* l) {
#if USE_ASYNC
    __builtin_amdgcn_global_load_async_to_lds_b128(
        (__attribute__((address_space(1))) v4i*)(void*)(g),
        (__attribute__((address_space(3))) v4i*)(void*)(l), 0, 0);
#else
    *(uint4*)l = *(const uint4*)g;
#endif
}

__device__ __forceinline__ void async_wait_all() {
#if USE_ASYNC
    __builtin_amdgcn_s_wait_asynccnt(0);
#endif
}

// ---------------------------------------------------------------------------
// Conversion / padding kernels (run once per layer; L2-resident streams)
// ---------------------------------------------------------------------------
// activations: out[row*padK + k] = bf16(in[row*K + k]) with zero padding
__global__ void cvt_act(const float* __restrict__ in, unsigned short* __restrict__ out,
                        int M, int K, int padK) {
    int k = blockIdx.x * blockDim.x + threadIdx.x;
    int row = blockIdx.y;
    if (k >= padK) return;
    float v = (row < M && k < K) ? in[(size_t)row * K + k] : 0.0f;
    out[(size_t)row * padK + k] = f2bf(v);
}

// weights: Wt[col*padK + k] = bf16(W[k*256 + col]) with zero padding  (transpose)
__global__ void cvt_wt(const float* __restrict__ W, unsigned short* __restrict__ Wt,
                       int K, int padK) {
    int k = blockIdx.x * blockDim.x + threadIdx.x;
    int col = blockIdx.y;
    if (k >= padK) return;
    float v = (k < K) ? W[(size_t)k * HIDD + col] : 0.0f;
    Wt[(size_t)col * padK + k] = f2bf(v);
}

// ---------------------------------------------------------------------------
// C[M,256] = A[M,padK] @ Wt^T (+ bias), bf16 WMMA, f32 accumulate.
// A: bf16 [MPAD, padK] row-major (zero padded).  Wt: bf16 [256, padK] (W^T).
// Block tile 128x128, 8 waves (4M x 2N), wave tile 32x64 = 2x4 WMMA tiles.
// Double-buffered LDS staging via async global->LDS b128 copies.
// ---------------------------------------------------------------------------
__global__ __launch_bounds__(256)
void wmma_gemm_bf16(const unsigned short* __restrict__ A,
                    const unsigned short* __restrict__ Wt,
                    const float* __restrict__ bias,
                    float* __restrict__ C, int M, int padK) {
    __shared__ __align__(16) unsigned short sA[2][128 * LDS_STRIDE];
    __shared__ __align__(16) unsigned short sB[2][128 * LDS_STRIDE];

    const int tid    = threadIdx.x;
    const int lane   = tid & 31;
    const int wave   = tid >> 5;
    const int wm     = wave & 3;
    const int wn     = wave >> 2;
    const int mblock = blockIdx.x * 128;
    const int nblock = blockIdx.y * 128;
    const int lane16 = lane & 15;
    const int kh     = lane >> 4;

    // each thread copies 2 b128 chunks per matrix per K-step (512 chunks total)
    const int c0 = tid, c1 = tid + 256;
    const int r0 = c0 >> 2, kc0 = (c0 & 3) * 8;
    const int r1 = c1 >> 2, kc1 = (c1 & 3) * 8;

    const unsigned short* Abase = A + (size_t)mblock * padK;
    const unsigned short* Bbase = Wt + (size_t)nblock * padK;

    v8f acc[2][4];
    const v8f vzero = {0.f, 0.f, 0.f, 0.f, 0.f, 0.f, 0.f, 0.f};
#pragma unroll
    for (int i = 0; i < 2; ++i)
#pragma unroll
        for (int j = 0; j < 4; ++j) acc[i][j] = vzero;

    const int nK = padK >> 5;

    // prologue: stage tile 0
    copy16(Abase + (size_t)r0 * padK + kc0, &sA[0][r0 * LDS_STRIDE + kc0]);
    copy16(Abase + (size_t)r1 * padK + kc1, &sA[0][r1 * LDS_STRIDE + kc1]);
    copy16(Bbase + (size_t)r0 * padK + kc0, &sB[0][r0 * LDS_STRIDE + kc0]);
    copy16(Bbase + (size_t)r1 * padK + kc1, &sB[0][r1 * LDS_STRIDE + kc1]);
    async_wait_all();
    __syncthreads();

    for (int kb = 0; kb < nK; ++kb) {
        const int cur = kb & 1;

        // issue async stage of next tile into the other buffer
        if (kb + 1 < nK) {
            const int nxt = cur ^ 1;
            const int k0n = (kb + 1) << 5;
            copy16(Abase + (size_t)r0 * padK + k0n + kc0, &sA[nxt][r0 * LDS_STRIDE + kc0]);
            copy16(Abase + (size_t)r1 * padK + k0n + kc1, &sA[nxt][r1 * LDS_STRIDE + kc1]);
            copy16(Bbase + (size_t)r0 * padK + k0n + kc0, &sB[nxt][r0 * LDS_STRIDE + kc0]);
            copy16(Bbase + (size_t)r1 * padK + k0n + kc1, &sB[nxt][r1 * LDS_STRIDE + kc1]);
        }

        // A 16-bit 16x32 fragment: lane half kh holds K = kh*8 + {0..7, 16..23}
        v16bf afrag[2];
#pragma unroll
        for (int tm = 0; tm < 2; ++tm) {
            const unsigned short* pa =
                &sA[cur][(wm * 32 + tm * 16 + lane16) * LDS_STRIDE + kh * 8];
            union { uint4 q[2]; v16bf v; } u;
            u.q[0] = *(const uint4*)(pa);
            u.q[1] = *(const uint4*)(pa + 16);
            afrag[tm] = u.v;
        }
        // B 16-bit 32x16 fragment: lane half kh holds K = kh*8 + {0..7, 16..23}
        v16bf bfrag[4];
#pragma unroll
        for (int tn = 0; tn < 4; ++tn) {
            const unsigned short* pb =
                &sB[cur][(wn * 64 + tn * 16 + lane16) * LDS_STRIDE + kh * 8];
            union { uint4 q[2]; v16bf v; } u;
            u.q[0] = *(const uint4*)(pb);
            u.q[1] = *(const uint4*)(pb + 16);
            bfrag[tn] = u.v;
        }

#pragma unroll
        for (int tm = 0; tm < 2; ++tm)
#pragma unroll
            for (int tn = 0; tn < 4; ++tn)
                acc[tm][tn] = __builtin_amdgcn_wmma_f32_16x16x32_bf16(
                    false, afrag[tm], false, bfrag[tn],
                    (short)0, acc[tm][tn], false, false);

        if (kb + 1 < nK) async_wait_all();   // own async copies landed
        __syncthreads();                     // everyone's copies landed
    }

    // store D: row = m + v + 8*kh, col = n + lane16
#pragma unroll
    for (int tm = 0; tm < 2; ++tm) {
#pragma unroll
        for (int tn = 0; tn < 4; ++tn) {
            int col  = nblock + wn * 64 + tn * 16 + lane16;
            float bv = bias ? bias[col] : 0.0f;
#pragma unroll
            for (int v = 0; v < 8; ++v) {
                int row = mblock + wm * 32 + tm * 16 + v + 8 * kh;
                if (row < M) C[(size_t)row * HIDD + col] = acc[tm][tn][v] + bv;
            }
        }
    }
}

// ---------------------------------------------------------------------------
// One wave32 per edge: agg[dst] += support[src] * w   (256 floats per row)
// ---------------------------------------------------------------------------
__global__ __launch_bounds__(256)
void scatter_edges(const float* __restrict__ sup, const int* __restrict__ src,
                   const int* __restrict__ dst, const float* __restrict__ ew,
                   float* __restrict__ agg, int nE) {
    int wid  = blockIdx.x * (blockDim.x >> 5) + (threadIdx.x >> 5);
    int lane = threadIdx.x & 31;
    if (wid >= nE) return;
    int s = src[wid], d = dst[wid];
    float w = ew[wid];
    const float4* sp = (const float4*)(sup + (size_t)s * HIDD);
    float* op = agg + (size_t)d * HIDD;
#pragma unroll
    for (int i = 0; i < 2; ++i) {
        int idx  = lane + i * 32;
        float4 v = sp[idx];
        float* o = op + idx * 4;
        unsafeAtomicAdd(o + 0, v.x * w);
        unsafeAtomicAdd(o + 1, v.y * w);
        unsafeAtomicAdd(o + 2, v.z * w);
        unsafeAtomicAdd(o + 3, v.w * w);
    }
}

// ---------------------------------------------------------------------------
// Elementwise helpers
// ---------------------------------------------------------------------------
__global__ void relu_k(float* __restrict__ p, int n) {
    int i = blockIdx.x * blockDim.x + threadIdx.x;
    if (i < n) p[i] = fmaxf(p[i], 0.0f);
}

__global__ void combine_k(float* __restrict__ h, const float* __restrict__ u, int n) {
    int i = blockIdx.x * blockDim.x + threadIdx.x;
    if (i < n) h[i] = (h[i] + fmaxf(u[i], 0.0f)) * 0.5f;
}

// ---------------------------------------------------------------------------
// Final 256->3 head (scalar; tiny) + its scatter
// ---------------------------------------------------------------------------
__global__ __launch_bounds__(256)
void final_layer(const float* __restrict__ h, const float* __restrict__ Wn,
                 const float* __restrict__ Wl, const float* __restrict__ b,
                 float* __restrict__ sup3, float* __restrict__ out3) {
    int r = blockIdx.x * blockDim.x + threadIdx.x;
    if (r >= NODES) return;
    const float* hr = h + (size_t)r * HIDD;
    float sn0 = 0.f, sn1 = 0.f, sn2 = 0.f;
    float sl0 = 0.f, sl1 = 0.f, sl2 = 0.f;
    for (int k = 0; k < HIDD; ++k) {
        float hv = hr[k];
        sn0 += hv * Wn[k * 3 + 0]; sn1 += hv * Wn[k * 3 + 1]; sn2 += hv * Wn[k * 3 + 2];
        sl0 += hv * Wl[k * 3 + 0]; sl1 += hv * Wl[k * 3 + 1]; sl2 += hv * Wl[k * 3 + 2];
    }
    sup3[(size_t)r * 3 + 0] = sn0; sup3[(size_t)r * 3 + 1] = sn1; sup3[(size_t)r * 3 + 2] = sn2;
    out3[(size_t)r * 3 + 0] = sl0 + b[0];
    out3[(size_t)r * 3 + 1] = sl1 + b[1];
    out3[(size_t)r * 3 + 2] = sl2 + b[2];
}

__global__ void scatter3(const float* __restrict__ sup3, const int* __restrict__ src,
                         const int* __restrict__ dst, const float* __restrict__ ew,
                         float* __restrict__ out3, int nE) {
    int e = blockIdx.x * blockDim.x + threadIdx.x;
    if (e >= nE) return;
    int s = src[e], d = dst[e];
    float w = ew[e];
    unsafeAtomicAdd(&out3[(size_t)d * 3 + 0], sup3[(size_t)s * 3 + 0] * w);
    unsafeAtomicAdd(&out3[(size_t)d * 3 + 1], sup3[(size_t)s * 3 + 1] * w);
    unsafeAtomicAdd(&out3[(size_t)d * 3 + 2], sup3[(size_t)s * 3 + 2] * w);
}

// ---------------------------------------------------------------------------
// Host orchestration
// ---------------------------------------------------------------------------
extern "C" void kernel_launch(void* const* d_in, const int* in_sizes, int n_in,
                              void* d_out, int out_size, void* d_ws, size_t ws_size,
                              hipStream_t stream) {
    const float* x      = (const float*)d_in[0];
    const float* edge_w = (const float*)d_in[1];
    const float* W1n    = (const float*)d_in[2];
    const float* W1l    = (const float*)d_in[3];
    const float* b1     = (const float*)d_in[4];
    const float* blkW1n = (const float*)d_in[5];
    const float* blkW1l = (const float*)d_in[6];
    const float* blkb1  = (const float*)d_in[7];
    const float* blkW2n = (const float*)d_in[8];
    const float* blkW2l = (const float*)d_in[9];
    const float* blkb2  = (const float*)d_in[10];
    const float* W2n    = (const float*)d_in[11];
    const float* W2l    = (const float*)d_in[12];
    const float* b2     = (const float*)d_in[13];
    const int*   esrc   = (const int*)d_in[14];
    const int*   edst   = (const int*)d_in[15];

    // d_out = [x_out (NODES*3) | x_hidden (NODES*256)]; x_hidden region is the
    // live h buffer, so no final copy is needed.
    float* x_out = (float*)d_out;
    float* h     = x_out + (size_t)NODES * OUTD;

    const size_t feat = (size_t)NODES * HIDD;
    float* sup = (float*)d_ws;
    float* t   = sup + feat;
    float* u   = t + feat;
    unsigned short* xbf = (unsigned short*)(u + feat);       // [MPAD, padK] bf16
    unsigned short* wnt = xbf + (size_t)MPAD * PADK1;        // [256, padK] bf16 (W^T)
    unsigned short* wlt = wnt + (size_t)HIDD * PADK1;

    dim3 gGrid(MPAD / 128, 2);
    dim3 gBlock(256);
    int  ewaves = (EDGES + 7) / 8;
    int  nfeat  = NODES * HIDD;
    int  eGrid  = (nfeat + 255) / 256;

    auto gconv = [&](const float* X, int K, int padK, const float* Wn, const float* Wl,
                     const float* bias, float* OUT, bool relu_out) {
        dim3 aGrid((padK + 255) / 256, MPAD);
        dim3 wGrid((padK + 255) / 256, HIDD);
        cvt_act<<<aGrid, 256, 0, stream>>>(X, xbf, NODES, K, padK);
        cvt_wt<<<wGrid, 256, 0, stream>>>(Wn, wnt, K, padK);
        cvt_wt<<<wGrid, 256, 0, stream>>>(Wl, wlt, K, padK);
        wmma_gemm_bf16<<<gGrid, gBlock, 0, stream>>>(xbf, wnt, nullptr, sup, NODES, padK);
        wmma_gemm_bf16<<<gGrid, gBlock, 0, stream>>>(xbf, wlt, bias, OUT, NODES, padK);
        scatter_edges<<<ewaves, 256, 0, stream>>>(sup, esrc, edst, edge_w, OUT, EDGES);
        if (relu_out) relu_k<<<eGrid, 256, 0, stream>>>(OUT, nfeat);
    };

    // layer 1: h = relu(gconv(x))   (K = 963 -> padded 992)
    gconv(x, INDIM, PADK1, W1n, W1l, b1, h, true);

    // 6 residual blocks
    for (int b = 0; b < NBLK; ++b) {
        const size_t wo = (size_t)b * HIDD * HIDD;
        const size_t bo = (size_t)b * HIDD;
        gconv(h, HIDD, HIDD, blkW1n + wo, blkW1l + wo, blkb1 + bo, t, true);
        gconv(t, HIDD, HIDD, blkW2n + wo, blkW2l + wo, blkb2 + bo, u, false);
        combine_k<<<eGrid, 256, 0, stream>>>(h, u, nfeat);   // h = (h + relu(u))/2
    }

    // head: x_out = gconv(h) with 256->3 weights (scalar path, tiny)
    final_layer<<<(NODES + 255) / 256, 256, 0, stream>>>(h, W2n, W2l, b2, sup, x_out);
    scatter3<<<(EDGES + 255) / 256, 256, 0, stream>>>(sup, esrc, edst, edge_w, x_out, EDGES);
}
